// Transformer_2147483648501
// MI455X (gfx1250) — matrix operand order repaired and checked by reading, not compiled, not run on hardware
//
#include <hip/hip_runtime.h>

typedef unsigned short u16;
typedef unsigned int   u32;
typedef __attribute__((ext_vector_type(16))) __bf16 v16bf;
typedef __attribute__((ext_vector_type(8)))  float  v8f;
typedef __attribute__((ext_vector_type(4)))  unsigned int u32x4;
typedef __attribute__((ext_vector_type(8)))  int i32x8;
typedef __attribute__((ext_vector_type(4)))  int i32x4;

union Frag16 { v16bf v; u32 u[8]; };     // A or B matrix fragment (16 bf16 / lane)
union FragC  { v8f  v; float f[8]; };    // C/D fragment (8 f32 / lane)

__device__ __forceinline__ u16 f2bf(float f) {
    u32 x = __float_as_uint(f);
    u32 r = x + 0x7FFFu + ((x >> 16) & 1u);   // round-to-nearest-even
    return (u16)(r >> 16);
}

#define BATCH 16
#define SEQ   1024
#define DIM   384
#define HEADS 6
#define DH    64
#define INNER 384
#define MLP   1536
#define DEPTH 6
#define ROWS  (BATCH * SEQ)   // 16384

// ---- workspace layout (u16 element offsets) ----
#define SZ_WQKV (DEPTH * DIM * 3 * INNER)
#define SZ_WO   (DEPTH * INNER * DIM)
#define SZ_W1   (DEPTH * DIM * MLP)
#define SZ_W2   (DEPTH * MLP * DIM)
#define SZ_H    (ROWS * DIM)
#define OFF_WQKV 0
#define OFF_WO   (OFF_WQKV + SZ_WQKV)
#define OFF_W1   (OFF_WO + SZ_WO)
#define OFF_W2   (OFF_W1 + SZ_W1)
#define OFF_H    (OFF_W2 + SZ_W2)
#define OFF_SCR  (OFF_H + SZ_H)
#define OFF_Q    (OFF_SCR)
#define OFF_KT   (OFF_Q  + BATCH * HEADS * SEQ * DH)
#define OFF_V    (OFF_KT + BATCH * HEADS * SEQ * DH)
#define OFF_O    (OFF_V  + BATCH * HEADS * SEQ * DH)
#define OFF_FF   (OFF_SCR)   // FF hidden aliases q/kT/v/o (never simultaneously live)

// ---- Tensor Data Mover: 2D bf16 tile load, D# per ISA cdna5 §8.3/8.4 ----
// tile_x elements contiguous (dim0), tile_y rows, row stride = stride_x elements.
// LDS fill is linear (tile_x*2 bytes per row, rows consecutive), padding off.
__device__ __forceinline__ void tdm_load_2d_bf16(unsigned lds_addr, const void* gptr,
                                                 unsigned tile_x, unsigned tile_y,
                                                 unsigned stride_x) {
    unsigned long long ga = (unsigned long long)(size_t)gptr;
    u32x4 g0;
    g0.x = 1u;                                            // count=1 (valid), user mode
    g0.y = lds_addr;                                      // lds_addr [63:32]
    g0.z = (u32)(ga & 0xffffffffu);                       // global_addr [95:64]
    g0.w = (u32)((ga >> 32) & 0x01ffffffu) | (2u << 30);  // addr[56:32] | type=2
    i32x8 g1;
    g1.s0 = (int)(1u << 16);                              // data_size=1 -> 2 bytes
    g1.s1 = (int)((tile_x & 0xffffu) << 16);              // tensor_dim0[15:0] (==tile)
    g1.s2 = (int)((tile_x >> 16) | ((tile_y & 0xffffu) << 16)); // dim0 hi | dim1 lo
    g1.s3 = (int)((tile_y >> 16) | ((tile_x & 0xffffu) << 16)); // dim1 hi | tile_dim0
    g1.s4 = (int)(tile_y & 0xffffu);                      // tile_dim1 (tile_dim2=0)
    g1.s5 = (int)stride_x;                                // tensor_dim0_stride[31:0]
    g1.s6 = 0;                                            // stride hi | dim1_stride lo
    g1.s7 = 0;
    i32x4 z4; z4.x = 0; z4.y = 0; z4.z = 0; z4.w = 0;     // groups 2/3 unused (2D)
    i32x8 z8; z8.s0 = 0; z8.s1 = 0; z8.s2 = 0; z8.s3 = 0;
    z8.s4 = 0; z8.s5 = 0; z8.s6 = 0; z8.s7 = 0;
    __builtin_amdgcn_tensor_load_to_lds(g0, g1, z4, z4, z8, 0);
}

__device__ __forceinline__ unsigned lds_off32(const void* p) {
    return (unsigned)(size_t)p;   // LDS generic ptr: low 32 bits = wave LDS offset
}

// ================= weight fp32 -> bf16 =================
__global__ void convert_f32_bf16(const float* __restrict__ s, u16* __restrict__ d, int n) {
    int i = blockIdx.x * blockDim.x + threadIdx.x;
    if (i < n) d[i] = f2bf(s[i]);
}

// ================= LayerNorm (fp32 in, bf16 out) =================
__global__ __launch_bounds__(128) void ln_bf16(const float* __restrict__ x,
                                               const float* __restrict__ g,
                                               const float* __restrict__ b,
                                               u16* __restrict__ out) {
    __shared__ float rs[128], rss[128];
    const int row = blockIdx.x, t = threadIdx.x;
    const float* xr = x + (size_t)row * DIM;
    float vals[3]; float s = 0.f, ss = 0.f;
#pragma unroll
    for (int i = 0; i < 3; i++) { float f = xr[t + i * 128]; vals[i] = f; s += f; ss += f * f; }
    rs[t] = s; rss[t] = ss; __syncthreads();
    for (int off = 64; off > 0; off >>= 1) {
        if (t < off) { rs[t] += rs[t + off]; rss[t] += rss[t + off]; }
        __syncthreads();
    }
    const float mean = rs[0] * (1.f / DIM);
    const float var  = rss[0] * (1.f / DIM) - mean * mean;
    const float inv  = rsqrtf(var + 1e-5f);
    u16* orow = out + (size_t)row * DIM;
#pragma unroll
    for (int i = 0; i < 3; i++) {
        int c = t + i * 128;
        orow[c] = f2bf((vals[i] - mean) * inv * g[c] + b[c]);
    }
}

// ================= tiled bf16 WMMA GEMM with TDM staging =================
// C[M,Nc] = A[M,K(lda)] * B[K,Nc(ldb)], block tile 128x128x32, 8 waves,
// each wave: 64x32 (4x2 grid of 16x16 f32 accumulators).
// A/B tiles are DMA'd global->LDS by the Tensor Data Mover (double-buffered):
// lead wave issues next tile's tensor_load_to_lds, all waves compute current
// buffer, then s_wait_tensorcnt 0 + barrier.
enum { EPI_QKV = 0, EPI_BIAS_RES = 1, EPI_GELU_BF16 = 2 };

template <int EPI>
__global__ __launch_bounds__(256) void gemm_bf16_wmma(
        const u16* __restrict__ A, int lda,
        const u16* __restrict__ Bw, int ldb,
        int K, int Nc,
        const float* __restrict__ bias,
        float* __restrict__ resid,          // EPI_BIAS_RES: read+write (residual stream)
        u16* __restrict__ outb,             // EPI_GELU_BF16 output
        u16* __restrict__ qp, u16* __restrict__ ktp, u16* __restrict__ vp) {
    __shared__ __align__(128) u16 As[2][128 * 32];
    __shared__ __align__(128) u16 Bs[2][32 * 128];
    const int tid  = threadIdx.x;
    const int lane = tid & 31, wave = tid >> 5;
    const int wm = (wave >> 2) * 64, wn = (wave & 3) * 32;
    const int bm = blockIdx.y * 128, bn = blockIdx.x * 128;
    const int r16 = lane & 15, grp = lane >> 4;
    // scalar (wave-uniform) predicate so the TDM issue sits behind an s_cbranch:
    // TENSOR ops ignore EXEC, so every wave reaching them would issue a DMA.
    const bool tdm_lead = (__builtin_amdgcn_readfirstlane(wave) == 0);

    FragC acc[4][2];
#pragma unroll
    for (int i = 0; i < 4; i++)
#pragma unroll
        for (int j = 0; j < 2; j++)
#pragma unroll
            for (int r = 0; r < 8; r++) acc[i][j].f[r] = 0.f;

    // ---- prologue: DMA k-tile 0 into buffer 0 ----
    if (tdm_lead) {
        tdm_load_2d_bf16(lds_off32(&As[0][0]), A + (size_t)bm * lda, 32, 128, (unsigned)lda);
        tdm_load_2d_bf16(lds_off32(&Bs[0][0]), Bw + bn, 128, 32, (unsigned)ldb);
        __builtin_amdgcn_s_wait_tensorcnt(0);
    }
    __syncthreads();

    const int nk = K / 32;
    for (int ki = 0; ki < nk; ki++) {
        const int buf = ki & 1;
        // ---- kick off next tile's DMA before computing this one ----
        if (tdm_lead && (ki + 1 < nk)) {
            const int k1 = (ki + 1) * 32;
            tdm_load_2d_bf16(lds_off32(&As[buf ^ 1][0]),
                             A + (size_t)bm * lda + k1, 32, 128, (unsigned)lda);
            tdm_load_2d_bf16(lds_off32(&Bs[buf ^ 1][0]),
                             Bw + (size_t)k1 * ldb + bn, 128, 32, (unsigned)ldb);
        }
        const u16* Asb = &As[buf][0];
        const u16* Bsb = &Bs[buf][0];
        // A fragment: lane holds row (lane&15); K packed in pairs per ISA 16-bit A layout
        Frag16 a[4], b[2];
#pragma unroll
        for (int i = 0; i < 4; i++) {
            const u16* base = Asb + (wm + i * 16 + r16) * 32 + grp * 8;
#pragma unroll
            for (int vv = 0; vv < 8; vv++) {
                int kb = (vv < 4) ? (vv * 2) : (16 + (vv - 4) * 2);
                a[i].u[vv] = *(const u32*)(base + kb);
            }
        }
        // B fragment: lane holds column (lane&15); K packed in pairs, same order as A
#pragma unroll
        for (int j = 0; j < 2; j++) {
            const u16* base = Bsb + grp * 8 * 128 + wn + j * 16 + r16;
#pragma unroll
            for (int vv = 0; vv < 8; vv++) {
                int kb = (vv < 4) ? (vv * 2) : (16 + (vv - 4) * 2);
                b[j].u[vv] = (u32)base[kb * 128] | ((u32)base[(kb + 1) * 128] << 16);
            }
        }
#pragma unroll
        for (int i = 0; i < 4; i++)
#pragma unroll
            for (int j = 0; j < 2; j++)
                acc[i][j].v = __builtin_amdgcn_wmma_f32_16x16x32_bf16(
                    false, a[i].v, false, b[j].v, (short)0, acc[i][j].v, false, false);
        // next tile's DMA must be complete before anyone reads buf^1
        if (tdm_lead) __builtin_amdgcn_s_wait_tensorcnt(0);
        __syncthreads();
    }

    // ---- epilogue ----
#pragma unroll
    for (int i = 0; i < 4; i++)
#pragma unroll
        for (int j = 0; j < 2; j++) {
            const int gn = bn + wn + j * 16 + r16;
#pragma unroll
            for (int r = 0; r < 8; r++) {
                const int gm = bm + wm + i * 16 + r + 8 * grp;
                float val = acc[i][j].f[r];
                if constexpr (EPI == EPI_BIAS_RES) {
                    size_t idx = (size_t)gm * Nc + gn;
                    resid[idx] = resid[idx] + bias[gn] + val;
                } else if constexpr (EPI == EPI_GELU_BF16) {
                    float t = val + bias[gn];
                    float gelu = 0.5f * t * (1.f + erff(t * 0.70710678118654752f));
                    outb[(size_t)gm * Nc + gn] = f2bf(gelu);
                } else { // EPI_QKV: scatter to Q [B,H,N,64], K^T [B,H,64,N], V [B,H,N,64]
                    int sec  = gn / INNER;
                    int c    = gn - sec * INNER;
                    int head = c >> 6, d = c & 63;
                    int bi   = gm >> 10, tok = gm & 1023;
                    size_t hb = (size_t)(bi * HEADS + head);
                    u16 bv = f2bf(val);
                    if (sec == 0)      qp [(hb * SEQ + tok) * DH + d]  = bv;
                    else if (sec == 1) ktp[(hb * DH + d) * SEQ + tok]  = bv;
                    else               vp [(hb * SEQ + tok) * DH + d]  = bv;
                }
            }
        }
}

// ================= flash-style LSA attention (WMMA) =================
// grid (N/64, B*H), 128 threads = 4 waves; wave owns 16 query rows.
__global__ __launch_bounds__(128) void attn_wmma(
        const u16* __restrict__ q, const u16* __restrict__ kt, const u16* __restrict__ v,
        u16* __restrict__ o, const float* __restrict__ temp, int layer) {
    __shared__ __align__(16) u16 Plds[4][16 * 32];
    const int lane = threadIdx.x & 31, wave = threadIdx.x >> 5;
    const int r16 = lane & 15, grp = lane >> 4;
    const int bh = blockIdx.y;
    const int bi = bh / HEADS, hd = bh - bi * HEADS;
    const int q0 = blockIdx.x * 64 + wave * 16;
    const u16* Q  = q  + (size_t)bh * SEQ * DH;
    const u16* KT = kt + (size_t)bh * DH * SEQ;
    const u16* V  = v  + (size_t)bh * SEQ * DH;
    const float scale = expf(temp[layer]);

    // Q fragments: 16 rows x 64 d, split into two 16x32 A fragments
    Frag16 qa[2];
#pragma unroll
    for (int hh = 0; hh < 2; hh++) {
        const u16* base = Q + (size_t)(q0 + r16) * DH + hh * 32 + grp * 8;
#pragma unroll
        for (int vv = 0; vv < 8; vv++) {
            int kb = (vv < 4) ? (vv * 2) : (16 + (vv - 4) * 2);
            qa[hh].u[vv] = *(const u32*)(base + kb);
        }
    }

    FragC oacc[4];
#pragma unroll
    for (int j = 0; j < 4; j++)
#pragma unroll
        for (int r = 0; r < 8; r++) oacc[j].f[r] = 0.f;
    float mrow[8], lrow[8];
#pragma unroll
    for (int r = 0; r < 8; r++) { mrow[r] = -1e30f; lrow[r] = 0.f; }

    for (int kb0 = 0; kb0 < SEQ; kb0 += 32) {
        // ---- S = Q K^T for 32 keys (two 16x16 tiles) ----
        FragC s[2];
#pragma unroll
        for (int t = 0; t < 2; t++) {
            const int key0 = kb0 + t * 16;
            Frag16 kf[2];
#pragma unroll
            for (int hh = 0; hh < 2; hh++) {
                const u16* base = KT + (size_t)(hh * 32 + grp * 8) * SEQ + key0 + r16;
#pragma unroll
                for (int vv = 0; vv < 8; vv++) {
                    int kb = (vv < 4) ? (vv * 2) : (16 + (vv - 4) * 2);
                    kf[hh].u[vv] = (u32)base[(size_t)kb * SEQ] | ((u32)base[(size_t)(kb + 1) * SEQ] << 16);
                }
            }
#pragma unroll
            for (int r = 0; r < 8; r++) s[t].f[r] = 0.f;
            s[t].v = __builtin_amdgcn_wmma_f32_16x16x32_bf16(
                false, qa[0].v, false, kf[0].v, (short)0, s[t].v, false, false);
            s[t].v = __builtin_amdgcn_wmma_f32_16x16x32_bf16(
                false, qa[1].v, false, kf[1].v, (short)0, s[t].v, false, false);
            const int kcol = key0 + r16;
#pragma unroll
            for (int r = 0; r < 8; r++) {
                float val = s[t].f[r] * scale;
                if ((q0 + r + 8 * grp) == kcol) val = -1e30f;   // LSA diagonal mask
                s[t].f[r] = val;
            }
        }
        // ---- online softmax (rows 0-7 in lanes 0-15, rows 8-15 in lanes 16-31) ----
        float mnew[8], corr[8];
#pragma unroll
        for (int r = 0; r < 8; r++) {
            float tm = fmaxf(s[0].f[r], s[1].f[r]);
#pragma unroll
            for (int m0 = 1; m0 < 16; m0 <<= 1) tm = fmaxf(tm, __shfl_xor(tm, m0, 32));
            mnew[r] = fmaxf(mrow[r], tm);
            corr[r] = expf(mrow[r] - mnew[r]);
        }
#pragma unroll
        for (int t = 0; t < 2; t++)
#pragma unroll
            for (int r = 0; r < 8; r++) s[t].f[r] = expf(s[t].f[r] - mnew[r]);
#pragma unroll
        for (int r = 0; r < 8; r++) {
            float ps = s[0].f[r] + s[1].f[r];
#pragma unroll
            for (int m0 = 1; m0 < 16; m0 <<= 1) ps += __shfl_xor(ps, m0, 32);
            lrow[r] = lrow[r] * corr[r] + ps;
            mrow[r] = mnew[r];
        }
#pragma unroll
        for (int j = 0; j < 4; j++)
#pragma unroll
            for (int r = 0; r < 8; r++) oacc[j].f[r] *= corr[r];
        // ---- repack P (C-layout f32) -> bf16 16x32 A-tile via per-wave LDS ----
#pragma unroll
        for (int t = 0; t < 2; t++)
#pragma unroll
            for (int r = 0; r < 8; r++)
                Plds[wave][(r + 8 * grp) * 32 + t * 16 + r16] = f2bf(s[t].f[r]);
        __syncthreads();
        Frag16 pa;
        {
            const u16* base = &Plds[wave][r16 * 32 + grp * 8];
#pragma unroll
            for (int vv = 0; vv < 8; vv++) {
                int kb = (vv < 4) ? (vv * 2) : (16 + (vv - 4) * 2);
                pa.u[vv] = *(const u32*)(base + kb);
            }
        }
        // ---- O += P V : 4 WMMAs over d ----
#pragma unroll
        for (int j = 0; j < 4; j++) {
            Frag16 vf;
            const u16* base = V + (size_t)(kb0 + grp * 8) * DH + j * 16 + r16;
#pragma unroll
            for (int vv = 0; vv < 8; vv++) {
                int kb = (vv < 4) ? (vv * 2) : (16 + (vv - 4) * 2);
                vf.u[vv] = (u32)base[kb * DH] | ((u32)base[(kb + 1) * DH] << 16);
            }
            oacc[j].v = __builtin_amdgcn_wmma_f32_16x16x32_bf16(
                false, pa.v, false, vf.v, (short)0, oacc[j].v, false, false);
        }
        __syncthreads();
    }
    // ---- finalize and store as [ROWS, INNER] bf16 (heads interleaved) ----
#pragma unroll
    for (int r = 0; r < 8; r++) {
        const int tok = q0 + r + 8 * grp;
        const size_t row = (size_t)(bi * SEQ + tok) * INNER + hd * DH;
        const float inv = 1.f / lrow[r];
#pragma unroll
        for (int j = 0; j < 4; j++) o[row + j * 16 + r16] = f2bf(oacc[j].f[r] * inv);
    }
}

// ================= launcher =================
extern "C" void kernel_launch(void* const* d_in, const int* in_sizes, int n_in,
                              void* d_out, int out_size, void* d_ws, size_t ws_size,
                              hipStream_t stream) {
    (void)in_sizes; (void)n_in; (void)out_size; (void)ws_size;
    const float* x_in  = (const float*)d_in[0];
    const float* ln1_g = (const float*)d_in[1];
    const float* ln1_b = (const float*)d_in[2];
    const float* qkv_w = (const float*)d_in[3];
    const float* temp  = (const float*)d_in[4];
    const float* out_w = (const float*)d_in[5];
    const float* out_b = (const float*)d_in[6];
    const float* ln2_g = (const float*)d_in[7];
    const float* ln2_b = (const float*)d_in[8];
    const float* ff_w1 = (const float*)d_in[9];
    const float* ff_b1 = (const float*)d_in[10];
    const float* ff_w2 = (const float*)d_in[11];
    const float* ff_b2 = (const float*)d_in[12];
    float* xr = (float*)d_out;            // residual stream lives in d_out (fp32)
    u16*   ws = (u16*)d_ws;

    (void)hipMemcpyAsync(xr, x_in, (size_t)ROWS * DIM * sizeof(float),
                         hipMemcpyDeviceToDevice, stream);

    convert_f32_bf16<<<(SZ_WQKV + 255) / 256, 256, 0, stream>>>(qkv_w, ws + OFF_WQKV, SZ_WQKV);
    convert_f32_bf16<<<(SZ_WO   + 255) / 256, 256, 0, stream>>>(out_w, ws + OFF_WO,   SZ_WO);
    convert_f32_bf16<<<(SZ_W1   + 255) / 256, 256, 0, stream>>>(ff_w1, ws + OFF_W1,   SZ_W1);
    convert_f32_bf16<<<(SZ_W2   + 255) / 256, 256, 0, stream>>>(ff_w2, ws + OFF_W2,   SZ_W2);

    for (int l = 0; l < DEPTH; l++) {
        // PreNorm + QKV projection
        ln_bf16<<<ROWS, 128, 0, stream>>>(xr, ln1_g + l * DIM, ln1_b + l * DIM, ws + OFF_H);
        gemm_bf16_wmma<EPI_QKV><<<dim3(9, 128), 256, 0, stream>>>(
            ws + OFF_H, DIM, ws + OFF_WQKV + (size_t)l * DIM * 3 * INNER, 3 * INNER,
            DIM, 3 * INNER, nullptr, nullptr, nullptr,
            ws + OFF_Q, ws + OFF_KT, ws + OFF_V);
        // LSA attention (flash, diagonal masked)
        attn_wmma<<<dim3(SEQ / 64, BATCH * HEADS), 128, 0, stream>>>(
            ws + OFF_Q, ws + OFF_KT, ws + OFF_V, ws + OFF_O, temp, l);
        // out projection + bias + residual
        gemm_bf16_wmma<EPI_BIAS_RES><<<dim3(3, 128), 256, 0, stream>>>(
            ws + OFF_O, INNER, ws + OFF_WO + (size_t)l * INNER * DIM, DIM,
            INNER, DIM, out_b + l * DIM, xr, nullptr, nullptr, nullptr, nullptr);
        // PreNorm + FF1 (GELU) + FF2 + residual
        ln_bf16<<<ROWS, 128, 0, stream>>>(xr, ln2_g + l * DIM, ln2_b + l * DIM, ws + OFF_H);
        gemm_bf16_wmma<EPI_GELU_BF16><<<dim3(12, 128), 256, 0, stream>>>(
            ws + OFF_H, DIM, ws + OFF_W1 + (size_t)l * DIM * MLP, MLP,
            DIM, MLP, ff_b1 + l * MLP, nullptr, ws + OFF_FF, nullptr, nullptr, nullptr);
        gemm_bf16_wmma<EPI_BIAS_RES><<<dim3(3, 128), 256, 0, stream>>>(
            ws + OFF_FF, MLP, ws + OFF_W2 + (size_t)l * MLP * DIM, DIM,
            MLP, DIM, ff_b2 + l * DIM, xr, nullptr, nullptr, nullptr, nullptr);
    }
}
